// Mamba_84868553769178
// MI455X (gfx1250) — matrix operand order repaired and checked by reading, not compiled, not run on hardware
//
#include <hip/hip_runtime.h>

// ---------------- problem constants (from reference) ----------------
#define D_MODEL   1024
#define D_STATE   16
#define D_CONV    4
#define D_INNER   2048        // EXPAND * D_MODEL
#define DT_RANK   64
#define BATCH     2
#define SEQLEN    4096
#define M_TOTAL   (BATCH*SEQLEN)      // 8192 rows for all GEMMs
#define XDBL_N    (DT_RANK + 2*D_STATE)   // 96
#define NCHUNK    64
#define CHUNK_LEN (SEQLEN/NCHUNK)     // 64
#define NCHAN     (BATCH*D_INNER)     // 4096 scan channels

// ---------------- types ----------------
typedef __attribute__((ext_vector_type(16))) __bf16        v16bf;
typedef __attribute__((ext_vector_type(2)))  __bf16        v2bf;
typedef __attribute__((ext_vector_type(8)))  float         v8f;
typedef __attribute__((ext_vector_type(8)))  unsigned int  v8u;
typedef __attribute__((ext_vector_type(4)))  float         v4f;

// packed fp32x2 -> bf16x2 (should lower to v_cvt_pk_bf16_f32-class op)
__device__ __forceinline__ unsigned pack_bf16(float x, float y) {
  v2bf p;
  p[0] = (__bf16)x;
  p[1] = (__bf16)y;
  return __builtin_bit_cast(unsigned, p);
}
__device__ __forceinline__ float silu(float x) {
  return x / (1.0f + __expf(-x));
}
__device__ __forceinline__ float softplus(float x) {
  return (x > 20.0f) ? x : log1pf(__expf(x));
}

// ================== WMMA GEMM: C[M,N] = A[M,K] * B[N,K]^T ==================
// Both operands K-contiguous (row-major A, row-major weight B) -> "NT" GEMM.
// fp32 in -> packed bf16 convert -> v_wmma_f32_16x16x32_bf16 -> fp32 out.
// Block tile 128x128, BK=32 (one wmma K-step), 256 threads = 8 wave32s in a
// 4x2 grid; each wave owns a 32x64 subtile (2x4 WMMAs per K-step).
// Double-buffered LDS + software-pipelined global loads: one barrier/K-step,
// global_load_b128 of tile k+1 issued before the 8-WMMA burst on tile k.
// M,K multiples of 128/32 (true for all four GEMMs); N is bounds-checked.
#define BM 128
#define BN 128
#define BK 32
#define LDS_S (BK + 8)   // bf16-element stride, pad to dodge bank conflicts

template<int EPI>   // 0 = plain store, 1 = softplus(acc + bias[n])
__global__ __launch_bounds__(256)
void gemm_bf16_wmma(const float* __restrict__ Ag, int lda,
                    const float* __restrict__ Bg, int ldb,
                    float* __restrict__ Cg, int ldc,
                    int N, int K,
                    const float* __restrict__ bias)
{
  __shared__ unsigned short As[2][BM * LDS_S];
  __shared__ unsigned short Bs[2][BN * LDS_S];

  const int tid  = threadIdx.x;
  const int lane = tid & 31;
  const int wave = tid >> 5;
  const int wm   = wave & 3;        // 0..3 -> 32-row slice
  const int wn   = wave >> 2;       // 0..1 -> 64-col slice
  const int bm0  = blockIdx.x * BM;
  const int bn0  = blockIdx.y * BN;

  // per-thread staging coordinates: 16 floats of A and of B as 4x float4
  int ar4[4], ac4[4];
#pragma unroll
  for (int i = 0; i < 4; ++i) {
    int idx = (i * 256 + tid) * 4;              // 0..4095
    ar4[i] = idx >> 5;                          // row 0..127
    ac4[i] = idx & 31;                          // col 0..28 step 4
  }

  v8f acc[2][4] = {};
  v4f areg[4], breg[4];

  // ---- prologue: fetch tile k0=0 and stage into buffer 0 ----
#pragma unroll
  for (int i = 0; i < 4; ++i) {
    areg[i] = *(const v4f*)(Ag + (size_t)(bm0 + ar4[i]) * lda + ac4[i]);
    v4f bv = {};
    if (bn0 + ar4[i] < N)
      bv = *(const v4f*)(Bg + (size_t)(bn0 + ar4[i]) * ldb + ac4[i]);
    breg[i] = bv;
  }
#pragma unroll
  for (int i = 0; i < 4; ++i) {
    *(unsigned*)&As[0][ar4[i] * LDS_S + ac4[i]]     = pack_bf16(areg[i].x, areg[i].y);
    *(unsigned*)&As[0][ar4[i] * LDS_S + ac4[i] + 2] = pack_bf16(areg[i].z, areg[i].w);
    *(unsigned*)&Bs[0][ar4[i] * LDS_S + ac4[i]]     = pack_bf16(breg[i].x, breg[i].y);
    *(unsigned*)&Bs[0][ar4[i] * LDS_S + ac4[i] + 2] = pack_bf16(breg[i].z, breg[i].w);
  }

  const int kb  = (lane >> 4) * 8;    // A frag: lanes 16-31 hold K=8-15/24-31
  const int kb2 = (lane >> 4) * 8;    // B frag: lanes 16-31 hold K=8-15/24-31
  int p = 0;

  for (int k0 = 0; k0 < K; k0 += BK, p ^= 1) {
    __syncthreads();
    const bool more = (k0 + BK) < K;

    // ---- issue next tile's global loads (latency hidden under WMMAs) ----
    if (more) {
#pragma unroll
      for (int i = 0; i < 4; ++i) {
        areg[i] = *(const v4f*)(Ag + (size_t)(bm0 + ar4[i]) * lda + k0 + BK + ac4[i]);
        v4f bv = {};
        if (bn0 + ar4[i] < N)
          bv = *(const v4f*)(Bg + (size_t)(bn0 + ar4[i]) * ldb + k0 + BK + ac4[i]);
        breg[i] = bv;
      }
    }

    // ---- load fragments per ISA 7.12.2 layouts, 8 WMMAs ----
    v8u afrag[2], bfrag[4];
#pragma unroll
    for (int i = 0; i < 2; ++i) {
      const unsigned short* arow =
          &As[p][((wm * 2 + i) * 16 + (lane & 15)) * LDS_S];
#pragma unroll
      for (int v = 0; v < 8; ++v) {
        int kk = kb + 2 * v + ((v >= 4) ? 8 : 0);   // K pairs 0-7 then 16-23
        afrag[i][v] = *(const unsigned*)(arow + kk);
      }
    }
#pragma unroll
    for (int j = 0; j < 4; ++j) {
      const unsigned short* brow =
          &Bs[p][((wn * 4 + j) * 16 + (lane & 15)) * LDS_S + kb2];
#pragma unroll
      for (int v = 0; v < 8; ++v)
        bfrag[j][v] = *(const unsigned*)(brow + 2 * v + ((v >= 4) ? 8 : 0));
    }
#pragma unroll
    for (int i = 0; i < 2; ++i)
#pragma unroll
      for (int j = 0; j < 4; ++j) {
        v16bf a = __builtin_bit_cast(v16bf, afrag[i]);
        v16bf b = __builtin_bit_cast(v16bf, bfrag[j]);
        acc[i][j] = __builtin_amdgcn_wmma_f32_16x16x32_bf16(
            false, a, false, b, (short)0, acc[i][j], false, false);
      }

    // ---- stage next tile into the other buffer ----
    if (more) {
      int q = p ^ 1;
#pragma unroll
      for (int i = 0; i < 4; ++i) {
        *(unsigned*)&As[q][ar4[i] * LDS_S + ac4[i]]     = pack_bf16(areg[i].x, areg[i].y);
        *(unsigned*)&As[q][ar4[i] * LDS_S + ac4[i] + 2] = pack_bf16(areg[i].z, areg[i].w);
        *(unsigned*)&Bs[q][ar4[i] * LDS_S + ac4[i]]     = pack_bf16(breg[i].x, breg[i].y);
        *(unsigned*)&Bs[q][ar4[i] * LDS_S + ac4[i] + 2] = pack_bf16(breg[i].z, breg[i].w);
      }
    }
  }

  // ---- epilogue: C/D layout -> VGPR e holds M = e (lanes<16) or 8+e ----
#pragma unroll
  for (int i = 0; i < 2; ++i)
#pragma unroll
    for (int j = 0; j < 4; ++j) {
      int mbase = bm0 + (wm * 2 + i) * 16 + (lane >> 4) * 8;
      int n     = bn0 + (wn * 4 + j) * 16 + (lane & 15);
      if (n < N) {
#pragma unroll
        for (int e = 0; e < 8; ++e) {
          float val = acc[i][j][e];
          if (EPI == 1) val = softplus(val + bias[n]);
          Cg[(size_t)(mbase + e) * ldc + n] = val;
        }
      }
    }
}

// ================== depthwise conv(4) + bias + SiLU ==================
// xc[b,t,d] = silu( sum_k x[b,t-3+k,d]*w[d,k] + cb[d] ), x = xz[:, :, 0:2048]
__global__ __launch_bounds__(256)
void conv_silu_kernel(const float* __restrict__ xz,
                      const float* __restrict__ cw,
                      const float* __restrict__ cb,
                      float* __restrict__ xc)
{
  size_t i = (size_t)blockIdx.x * blockDim.x + threadIdx.x;
  if (i >= (size_t)M_TOTAL * D_INNER) return;
  int d = (int)(i % D_INNER);
  size_t bt = i / D_INNER;
  int t = (int)(bt % SEQLEN);
  size_t brow = bt - t;                 // b*SEQLEN
  float s = cb[d];
#pragma unroll
  for (int k = 0; k < D_CONV; ++k) {
    int tt = t - (D_CONV - 1) + k;
    if (tt >= 0)
      s += xz[(brow + tt) * (size_t)(2 * D_INNER) + d] * cw[d * D_CONV + k];
  }
  xc[i] = silu(s);
}

// ================== chunked selective scan ==================
// h_t[n] = dA_t[n]*h_{t-1}[n] + dt_t*x_t*B_t[n]   (diagonal linear recurrence)
// pass1: per chunk compute P = prod dA, S = h_end with h_in = 0
__global__ __launch_bounds__(256)
void scan_pass1(const float* __restrict__ dt, const float* __restrict__ xc,
                const float* __restrict__ xdbl, const float* __restrict__ A_log,
                float* __restrict__ P, float* __restrict__ S)
{
  const int c    = blockIdx.x;                       // chunk
  const int cd   = blockIdx.y * 256 + threadIdx.x;   // channel b*D_INNER+d
  const int lane = threadIdx.x & 31;
  const int b    = cd / D_INNER, d = cd % D_INNER;

  float Ac[D_STATE];
#pragma unroll
  for (int n = 0; n < D_STATE; ++n) Ac[n] = -__expf(A_log[d * D_STATE + n]);

  float p[D_STATE], s[D_STATE];
#pragma unroll
  for (int n = 0; n < D_STATE; ++n) { p[n] = 1.0f; s[n] = 0.0f; }

  const int t0 = c * CHUNK_LEN;
  for (int tt = 0; tt < CHUNK_LEN; ++tt) {
    size_t row = (size_t)b * SEQLEN + (t0 + tt);
    float dtv = dt[row * D_INNER + d];
    float xv  = xc[row * D_INNER + d];
    float bc  = xdbl[row * XDBL_N + DT_RANK + lane];  // B||C, 32 floats shared
    float dtx = dtv * xv;
#pragma unroll
    for (int n = 0; n < D_STATE; ++n) {
      float Bn = __shfl(bc, n, 32);
      float dA = __expf(dtv * Ac[n]);
      p[n] *= dA;
      s[n] = dA * s[n] + dtx * Bn;
    }
  }
  size_t o = ((size_t)c * NCHAN + cd) * D_STATE;
#pragma unroll
  for (int n = 0; n < D_STATE; ++n) { P[o + n] = p[n]; S[o + n] = s[n]; }
}

// pass2 of the scan tree: tiny sequential combine over 64 chunks
__global__ __launch_bounds__(256)
void scan_combine(const float* __restrict__ P, const float* __restrict__ S,
                  float* __restrict__ Hinit)
{
  int idx = blockIdx.x * blockDim.x + threadIdx.x;   // cd*16+n, 65536 total
  float h = 0.0f;
  for (int c = 0; c < NCHUNK; ++c) {
    size_t o = (size_t)c * (NCHAN * D_STATE) + idx;
    Hinit[o] = h;
    h = P[o] * h + S[o];
  }
}

// pass3: redo chunks with correct h_init, compute y, fuse (y+x*D)*silu(z)
__global__ __launch_bounds__(256)
void scan_pass2(const float* __restrict__ dt, const float* __restrict__ xc,
                const float* __restrict__ xdbl, const float* __restrict__ A_log,
                const float* __restrict__ Hinit, const float* __restrict__ Dp,
                const float* __restrict__ xz, float* __restrict__ G)
{
  const int c    = blockIdx.x;
  const int cd   = blockIdx.y * 256 + threadIdx.x;
  const int lane = threadIdx.x & 31;
  const int b    = cd / D_INNER, d = cd % D_INNER;

  float Ac[D_STATE];
#pragma unroll
  for (int n = 0; n < D_STATE; ++n) Ac[n] = -__expf(A_log[d * D_STATE + n]);

  float h[D_STATE];
  size_t ho = ((size_t)c * NCHAN + cd) * D_STATE;
#pragma unroll
  for (int n = 0; n < D_STATE; ++n) h[n] = Hinit[ho + n];

  const float dpar = Dp[d];
  const int t0 = c * CHUNK_LEN;
  for (int tt = 0; tt < CHUNK_LEN; ++tt) {
    size_t row = (size_t)b * SEQLEN + (t0 + tt);
    float dtv = dt[row * D_INNER + d];
    float xv  = xc[row * D_INNER + d];
    float bc  = xdbl[row * XDBL_N + DT_RANK + lane];
    float dtx = dtv * xv;
    float y = 0.0f;
#pragma unroll
    for (int n = 0; n < D_STATE; ++n) {
      float Bn = __shfl(bc, n, 32);
      float Cn = __shfl(bc, D_STATE + n, 32);
      float dA = __expf(dtv * Ac[n]);
      h[n] = dA * h[n] + dtx * Bn;
      y += h[n] * Cn;
    }
    y += xv * dpar;
    float zv = xz[row * (size_t)(2 * D_INNER) + D_INNER + d];
    G[row * D_INNER + d] = y * silu(zv);
  }
}

// ================== launcher ==================
extern "C" void kernel_launch(void* const* d_in, const int* in_sizes, int n_in,
                              void* d_out, int out_size, void* d_ws, size_t ws_size,
                              hipStream_t stream) {
  const float* hidden   = (const float*)d_in[0];
  const float* in_w     = (const float*)d_in[1];
  const float* conv_w   = (const float*)d_in[2];
  const float* conv_b   = (const float*)d_in[3];
  const float* xproj_w  = (const float*)d_in[4];
  const float* dtproj_w = (const float*)d_in[5];
  const float* dtproj_b = (const float*)d_in[6];
  const float* A_log    = (const float*)d_in[7];
  const float* D_param  = (const float*)d_in[8];
  const float* out_w    = (const float*)d_in[9];
  float* out = (float*)d_out;

  // workspace carve-up (floats); total ~97.3M floats = ~389 MB
  float* ws   = (float*)d_ws;
  float* xz   = ws;                                   // 8192*4096
  float* xc   = xz   + (size_t)M_TOTAL * 2 * D_INNER; // 8192*2048
  float* xdbl = xc   + (size_t)M_TOTAL * D_INNER;     // 8192*96
  float* dt   = xdbl + (size_t)M_TOTAL * XDBL_N;      // 8192*2048
  float* G    = dt   + (size_t)M_TOTAL * D_INNER;     // 8192*2048
  float* P    = G    + (size_t)M_TOTAL * D_INNER;     // 64*4096*16
  float* S    = P    + (size_t)NCHUNK * NCHAN * D_STATE;
  float* H    = S    + (size_t)NCHUNK * NCHAN * D_STATE;

  dim3 blk(256);

  // 1) xz = hidden @ in_proj_w^T   (M=8192, N=4096, K=1024)
  gemm_bf16_wmma<0><<<dim3(M_TOTAL / BM, (2 * D_INNER) / BN), blk, 0, stream>>>(
      hidden, D_MODEL, in_w, D_MODEL, xz, 2 * D_INNER, 2 * D_INNER, D_MODEL, nullptr);

  // 2) depthwise conv + bias + SiLU
  conv_silu_kernel<<<dim3((M_TOTAL * D_INNER) / 256), blk, 0, stream>>>(
      xz, conv_w, conv_b, xc);

  // 3) x_dbl = xc @ x_proj_w^T     (N=96, bounds-checked, single N-tile)
  gemm_bf16_wmma<0><<<dim3(M_TOTAL / BM, (XDBL_N + BN - 1) / BN), blk, 0, stream>>>(
      xc, D_INNER, xproj_w, D_INNER, xdbl, XDBL_N, XDBL_N, D_INNER, nullptr);

  // 4) dt = softplus(x_dbl[:, :64] @ dt_proj_w^T + b)  (K=64, fused epilogue)
  gemm_bf16_wmma<1><<<dim3(M_TOTAL / BM, D_INNER / BN), blk, 0, stream>>>(
      xdbl, XDBL_N, dtproj_w, DT_RANK, dt, D_INNER, D_INNER, DT_RANK, dtproj_b);

  // 5-7) chunked parallel selective scan
  scan_pass1<<<dim3(NCHUNK, NCHAN / 256), blk, 0, stream>>>(
      dt, xc, xdbl, A_log, P, S);
  scan_combine<<<dim3((NCHAN * D_STATE) / 256), blk, 0, stream>>>(P, S, H);
  scan_pass2<<<dim3(NCHUNK, NCHAN / 256), blk, 0, stream>>>(
      dt, xc, xdbl, A_log, H, D_param, xz, G);

  // 8) out = G @ out_proj_w^T      (M=8192, N=1024, K=2048)
  gemm_bf16_wmma<0><<<dim3(M_TOTAL / BM, D_MODEL / BN), blk, 0, stream>>>(
      G, D_INNER, out_w, D_INNER, out, D_MODEL, D_MODEL, D_INNER, nullptr);
}
